// WikipediaMoeLayer_47639777247360
// MI455X (gfx1250) — matrix operand order repaired and checked by reading, not compiled, not run on hardware
//
#include <hip/hip_runtime.h>

// Problem constants (from reference): B=4, T=2048 -> 8192 tokens
#define NTOK 8192
#define CDIM 1024
#define HDIM 2048
#define NEXP 8

typedef __attribute__((ext_vector_type(16))) __bf16        v16bf;
typedef __attribute__((ext_vector_type(8)))  float         v8f;
typedef __attribute__((ext_vector_type(4)))  float         v4f;
typedef __attribute__((ext_vector_type(4)))  unsigned int  v4u;

union FragBF { v4u q[2]; v16bf v; };

__device__ __forceinline__ unsigned int f2bf1(float f) {
  unsigned int u = __float_as_uint(f);
  return (u + 0x7FFFu + ((u >> 16) & 1u)) >> 16;   // round-to-nearest-even bf16
}
__device__ __forceinline__ unsigned int pk2(float lo, float hi) {
  return f2bf1(lo) | (f2bf1(hi) << 16);
}

// ---------------- zero output + routing counts + dummy zero row ----------------
__global__ void moe_zero(v4f* __restrict__ out4, int* __restrict__ counts,
                         v4u* __restrict__ zrow4) {
  size_t i = (size_t)blockIdx.x * blockDim.x + threadIdx.x;
  v4f z = {0.f, 0.f, 0.f, 0.f};
  out4[i] = z;
  if (blockIdx.x == 0 && threadIdx.x < NEXP) counts[threadIdx.x] = 0;
  if (blockIdx.x == 1 && threadIdx.x < 128) {   // CDIM bf16 = 2KB = 128 x 16B
    v4u zu = {0u, 0u, 0u, 0u};
    zrow4[threadIdx.x] = zu;
  }
}

// ---------------- one-time f32 -> bf16 conversion (8 elts/thread) ----------------
__global__ void cvt_bf16(const float* __restrict__ src, unsigned short* __restrict__ dst) {
  size_t i = ((size_t)blockIdx.x * blockDim.x + threadIdx.x) * 8;
  const v4f* s = (const v4f*)(src + i);
  v4f f0 = s[0], f1 = s[1];
  v4u r;
  r.x = pk2(f0.x, f0.y); r.y = pk2(f0.z, f0.w);
  r.z = pk2(f1.x, f1.y); r.w = pk2(f1.z, f1.w);
  *(v4u*)(dst + i) = r;
}

// ---------------- gating: logits, top-2 softmax, routing lists ----------------
__global__ void moe_gate(const float* __restrict__ x, const float* __restrict__ gw,
                         float* __restrict__ wfull, int* __restrict__ counts,
                         int* __restrict__ lists) {
  int lane  = threadIdx.x & 31;
  int token = blockIdx.x * 8 + (threadIdx.x >> 5);   // one wave per token
  float acc[NEXP];
#pragma unroll
  for (int e = 0; e < NEXP; ++e) acc[e] = 0.f;
  const float* xr = x + (size_t)token * CDIM;
  for (int c = lane; c < CDIM; c += 32) {
    float xv = xr[c];
#pragma unroll
    for (int e = 0; e < NEXP; ++e) acc[e] += xv * gw[e * CDIM + c];
  }
#pragma unroll
  for (int e = 0; e < NEXP; ++e)
    for (int off = 16; off > 0; off >>= 1)
      acc[e] += __shfl_xor(acc[e], off, 32);
  if (lane == 0) {
    int e0 = 0; float v0 = acc[0];
#pragma unroll
    for (int e = 1; e < NEXP; ++e) if (acc[e] > v0) { v0 = acc[e]; e0 = e; }
    int e1 = -1; float v1 = -3.4e38f;
#pragma unroll
    for (int e = 0; e < NEXP; ++e) if (e != e0 && acc[e] > v1) { v1 = acc[e]; e1 = e; }
    float t  = __expf(v1 - v0);
    float d  = 1.f / (1.f + t);
    wfull[token * NEXP + e0] = d;
    wfull[token * NEXP + e1] = t * d;
    int p0 = atomicAdd(&counts[e0], 1); lists[e0 * NTOK + p0] = token;
    int p1 = atomicAdd(&counts[e1], 1); lists[e1 * NTOK + p1] = token;
  }
}

// ---------------- pad per-expert lists to a multiple of 128 ----------------
__global__ void moe_pad(const int* __restrict__ counts, int* __restrict__ lists) {
  int e = blockIdx.x;
  int cnt = counts[e];
  int padded = (cnt + 127) & ~127;
  for (int i = cnt + (int)threadIdx.x; i < padded; i += blockDim.x)
    lists[e * NTOK + i] = -1;
}

// ---- fused gate/up GEMM (all-bf16 operands, double-buffered LDS) + SiLU*up ----
// Block tile M=128 x N=64, K-step 32. 8 waves as 4x2, wave tile 32x32.
__global__ __launch_bounds__(256) void moe_up_gemm(
    const unsigned short* __restrict__ xb, const unsigned short* __restrict__ wgb,
    const unsigned short* __restrict__ wub, const int* __restrict__ lists,
    const int* __restrict__ counts, const unsigned short* __restrict__ zrow,
    unsigned short* __restrict__ hbuf, int e) {
  int padded = (counts[e] + 127) & ~127;
  if ((int)blockIdx.x * 128 >= padded) return;

  __shared__ __align__(16) unsigned int As[2][128 * 16];  // 128 x 32 bf16, 2 stages
  __shared__ __align__(16) unsigned int Bs[2][128 * 16];  // rows 0-63: Wg, 64-127: Wu

  const int tid = threadIdx.x;
  const int lane = tid & 31, wid = tid >> 5;
  const int waveM = wid >> 1, waveN = wid & 1;
  const int l15 = lane & 15, hi = lane >> 4, rh8 = hi * 8;

  // loader roles: each thread copies one A half-row and one B half-row (b128 x2 each)
  const int lr = tid >> 1, lch = tid & 1;
  const int token = lists[e * NTOK + blockIdx.x * 128 + lr];
  // padded rows (-1) read a zeroed dummy row: unconditional loads, no divergence
  const unsigned short* aBase =
      (token >= 0) ? xb + (size_t)token * CDIM + lch * 16 : zrow + lch * 16;
  const unsigned short* wsel = (lr & 64) ? wub : wgb;
  const unsigned short* bBase =
      wsel + (size_t)(blockIdx.y * 64 + (lr & 63)) * CDIM + lch * 16;
  const int ldsOff = lr * 16 + lch * 8;

  v8f accg[2][2], accu[2][2];
#pragma unroll
  for (int mi = 0; mi < 2; ++mi)
#pragma unroll
    for (int ni = 0; ni < 2; ++ni) {
      v8f z = {0.f,0.f,0.f,0.f,0.f,0.f,0.f,0.f};
      accg[mi][ni] = z; accu[mi][ni] = z;
    }

  // prologue: stage k-block 0 into buffer 0
  {
    const v4u* p = (const v4u*)aBase;
    const v4u* q = (const v4u*)bBase;
    v4u a0 = p[0], a1 = p[1];
    v4u b0 = q[0], b1 = q[1];
    *(v4u*)&As[0][ldsOff] = a0; *(v4u*)&As[0][ldsOff + 4] = a1;
    *(v4u*)&Bs[0][ldsOff] = b0; *(v4u*)&Bs[0][ldsOff + 4] = b1;
  }

  for (int kb = 0; kb < CDIM; kb += 32) {
    const int cur = (kb >> 5) & 1;
    const bool hasNext = (kb + 32) < CDIM;
    v4u na0, na1, nb0, nb1;
    if (hasNext) {  // issue next-stage global loads before the barrier
      const v4u* p = (const v4u*)(aBase + kb + 32);
      const v4u* q = (const v4u*)(bBase + kb + 32);
      na0 = p[0]; na1 = p[1];
      nb0 = q[0]; nb1 = q[1];
    }
    __syncthreads();   // current stage visible to all

    FragBF af[2], bg[2], bu[2];
#pragma unroll
    for (int mi = 0; mi < 2; ++mi) {
      int r = waveM * 32 + mi * 16 + l15;
      af[mi].q[0] = *(const v4u*)&As[cur][r * 16 + hi * 8];      // K 0..7 / 16..23
      af[mi].q[1] = *(const v4u*)&As[cur][r * 16 + hi * 8 + 4];  // K 8..15 / 24..31
    }
#pragma unroll
    for (int ni = 0; ni < 2; ++ni) {
      int n = waveN * 32 + ni * 16 + l15;
      bg[ni].q[0] = *(const v4u*)&Bs[cur][n * 16 + hi * 8];
      bg[ni].q[1] = *(const v4u*)&Bs[cur][n * 16 + hi * 8 + 4];
      bu[ni].q[0] = *(const v4u*)&Bs[cur][(n + 64) * 16 + hi * 8];
      bu[ni].q[1] = *(const v4u*)&Bs[cur][(n + 64) * 16 + hi * 8 + 4];
    }
#pragma unroll
    for (int mi = 0; mi < 2; ++mi)
#pragma unroll
      for (int ni = 0; ni < 2; ++ni) {
        accg[mi][ni] = __builtin_amdgcn_wmma_f32_16x16x32_bf16(
            false, af[mi].v, false, bg[ni].v, (short)0, accg[mi][ni], false, false);
        accu[mi][ni] = __builtin_amdgcn_wmma_f32_16x16x32_bf16(
            false, af[mi].v, false, bu[ni].v, (short)0, accu[mi][ni], false, false);
      }
    if (hasNext) {   // fill the other buffer; next loop-top barrier publishes it
      const int bufn = cur ^ 1;
      *(v4u*)&As[bufn][ldsOff] = na0; *(v4u*)&As[bufn][ldsOff + 4] = na1;
      *(v4u*)&Bs[bufn][ldsOff] = nb0; *(v4u*)&Bs[bufn][ldsOff + 4] = nb1;
    }
  }

  const int rowbase = blockIdx.x * 128 + waveM * 32 + rh8;
  const int colbase = blockIdx.y * 64 + waveN * 32 + l15;
#pragma unroll
  for (int mi = 0; mi < 2; ++mi)
#pragma unroll
    for (int ni = 0; ni < 2; ++ni)
#pragma unroll
      for (int r = 0; r < 8; ++r) {
        float g = accg[mi][ni][r];
        float u = accu[mi][ni][r];
        float s = g / (1.f + __expf(-g));   // SiLU(g)
        hbuf[(size_t)(rowbase + mi * 16 + r) * HDIM + (colbase + ni * 16)] =
            (unsigned short)f2bf1(s * u);
      }
}

// ---- down GEMM: y = h * Wd^T (bf16), scaled atomicAdd into out ----
__global__ __launch_bounds__(256) void moe_down_gemm(
    const unsigned short* __restrict__ hbuf, const unsigned short* __restrict__ wdb,
    const int* __restrict__ lists, const int* __restrict__ counts,
    const float* __restrict__ wfull, float* __restrict__ out, int e) {
  int padded = (counts[e] + 127) & ~127;
  if ((int)blockIdx.x * 128 >= padded) return;

  __shared__ __align__(16) unsigned int As[2][128 * 16];  // h tile
  __shared__ __align__(16) unsigned int Bd[2][64 * 16];   // Wd^T tile

  const int tid = threadIdx.x;
  const int lane = tid & 31, wid = tid >> 5;
  const int waveM = wid >> 1, waveN = wid & 1;
  const int l15 = lane & 15, hi = lane >> 4, rh8 = hi * 8;

  const int lr = tid >> 1, lch = tid & 1;
  const unsigned short* aBase =
      hbuf + (size_t)(blockIdx.x * 128 + lr) * HDIM + lch * 16;
  const unsigned short* bBase =
      wdb + (size_t)(blockIdx.y * 64 + (lr & 63)) * HDIM + lch * 16;  // tid<128 only
  const int ldsOff = lr * 16 + lch * 8;
  const bool bLoader = tid < 128;

  v8f acc[2][2];
#pragma unroll
  for (int mi = 0; mi < 2; ++mi)
#pragma unroll
    for (int ni = 0; ni < 2; ++ni) {
      v8f z = {0.f,0.f,0.f,0.f,0.f,0.f,0.f,0.f};
      acc[mi][ni] = z;
    }

  {
    const v4u* p = (const v4u*)aBase;
    v4u a0 = p[0], a1 = p[1];
    *(v4u*)&As[0][ldsOff] = a0; *(v4u*)&As[0][ldsOff + 4] = a1;
    if (bLoader) {
      const v4u* q = (const v4u*)bBase;
      v4u b0 = q[0], b1 = q[1];
      *(v4u*)&Bd[0][ldsOff] = b0; *(v4u*)&Bd[0][ldsOff + 4] = b1;
    }
  }

  for (int kb = 0; kb < HDIM; kb += 32) {
    const int cur = (kb >> 5) & 1;
    const bool hasNext = (kb + 32) < HDIM;
    v4u na0, na1, nb0, nb1;
    if (hasNext) {
      const v4u* p = (const v4u*)(aBase + kb + 32);
      na0 = p[0]; na1 = p[1];
      if (bLoader) {
        const v4u* q = (const v4u*)(bBase + kb + 32);
        nb0 = q[0]; nb1 = q[1];
      }
    }
    __syncthreads();

    FragBF af[2], bf_[2];
#pragma unroll
    for (int mi = 0; mi < 2; ++mi) {
      int r = waveM * 32 + mi * 16 + l15;
      af[mi].q[0] = *(const v4u*)&As[cur][r * 16 + hi * 8];
      af[mi].q[1] = *(const v4u*)&As[cur][r * 16 + hi * 8 + 4];
    }
#pragma unroll
    for (int ni = 0; ni < 2; ++ni) {
      int n = waveN * 32 + ni * 16 + l15;
      bf_[ni].q[0] = *(const v4u*)&Bd[cur][n * 16 + hi * 8];
      bf_[ni].q[1] = *(const v4u*)&Bd[cur][n * 16 + hi * 8 + 4];
    }
#pragma unroll
    for (int mi = 0; mi < 2; ++mi)
#pragma unroll
      for (int ni = 0; ni < 2; ++ni)
        acc[mi][ni] = __builtin_amdgcn_wmma_f32_16x16x32_bf16(
            false, af[mi].v, false, bf_[ni].v, (short)0, acc[mi][ni], false, false);
    if (hasNext) {
      const int bufn = cur ^ 1;
      *(v4u*)&As[bufn][ldsOff] = na0; *(v4u*)&As[bufn][ldsOff + 4] = na1;
      if (bLoader) {
        *(v4u*)&Bd[bufn][ldsOff] = nb0; *(v4u*)&Bd[bufn][ldsOff + 4] = nb1;
      }
    }
  }

  const int rowbase = blockIdx.x * 128 + waveM * 32 + rh8;
  const int colbase = blockIdx.y * 64 + waveN * 32 + l15;
#pragma unroll
  for (int mi = 0; mi < 2; ++mi)
#pragma unroll
    for (int r = 0; r < 8; ++r) {
      int m = rowbase + mi * 16 + r;
      int token = lists[e * NTOK + m];
      if (token >= 0) {
        float w = wfull[token * NEXP + e];
#pragma unroll
        for (int ni = 0; ni < 2; ++ni)
          atomicAdd(&out[(size_t)token * CDIM + (colbase + ni * 16)],
                    acc[mi][ni][r] * w);   // exactly 2 commutative adds/elt -> deterministic
      }
    }
}

extern "C" void kernel_launch(void* const* d_in, const int* in_sizes, int n_in,
                              void* d_out, int out_size, void* d_ws, size_t ws_size,
                              hipStream_t stream) {
  (void)in_sizes; (void)n_in; (void)out_size; (void)ws_size;
  const float* x  = (const float*)d_in[0];   // [B,T,C]
  const float* gw = (const float*)d_in[1];   // [E,C]
  const float* Wg = (const float*)d_in[2];   // [E,H,C]
  const float* Wu = (const float*)d_in[3];   // [E,H,C]
  const float* Wd = (const float*)d_in[4];   // [E,C,H]
  float* out = (float*)d_out;

  char* ws = (char*)d_ws;
  float*          wfull  = (float*)ws;                                      // 256 KB
  int*            lists  = (int*)(ws + (256u << 10));                       // 256 KB
  int*            counts = (int*)(ws + (512u << 10));                       // 32 B
  unsigned short* zrow   = (unsigned short*)(ws + (768u << 10));            // 2 KB zeros
  unsigned short* xb     = (unsigned short*)(ws + (size_t)(1u << 20));      // 16 MB bf16
  unsigned short* wgb    = (unsigned short*)(ws + (size_t)17 * (1u << 20)); // 4 MB
  unsigned short* wub    = (unsigned short*)(ws + (size_t)21 * (1u << 20)); // 4 MB
  unsigned short* wdb    = (unsigned short*)(ws + (size_t)25 * (1u << 20)); // 4 MB
  unsigned short* hbuf   = (unsigned short*)(ws + (size_t)29 * (1u << 20)); // 32 MB

  moe_zero<<<(NTOK * CDIM) / (256 * 4), 256, 0, stream>>>((v4f*)out, counts, (v4u*)zrow);
  cvt_bf16<<<(NTOK * CDIM) / (256 * 8), 256, 0, stream>>>(x, xb);
  moe_gate<<<NTOK / 8, 256, 0, stream>>>(x, gw, wfull, counts, lists);
  moe_pad<<<NEXP, 128, 0, stream>>>(counts, lists);

  const int wElems = HDIM * CDIM;                     // 2,097,152 per matrix
  for (int e = 0; e < NEXP; ++e) {
    cvt_bf16<<<wElems / (256 * 8), 256, 0, stream>>>(Wg + (size_t)e * wElems, wgb);
    cvt_bf16<<<wElems / (256 * 8), 256, 0, stream>>>(Wu + (size_t)e * wElems, wub);
    cvt_bf16<<<wElems / (256 * 8), 256, 0, stream>>>(Wd + (size_t)e * wElems, wdb);
    moe_up_gemm<<<dim3(NTOK / 128, HDIM / 64), 256, 0, stream>>>(
        xb, wgb, wub, lists, counts, zrow, hbuf, e);
    moe_down_gemm<<<dim3(NTOK / 128, CDIM / 64), 256, 0, stream>>>(
        hbuf, wdb, lists, counts, wfull, out, e);
  }
}
